// DualPtrRNNDecoder_43078521979679
// MI455X (gfx1250) — matrix operand order repaired and checked by reading, not compiled, not run on hardware
//
#include <hip/hip_runtime.h>
#include <cstdint>
#include <cstddef>

#define Bv 64
#define Tv 64
#define DINv 512
#define Hv 512
#define SCv 512
#define SQv 128
#define NWG 64
#define NTHR 256

typedef __attribute__((ext_vector_type(16))) __bf16 v16bf;
typedef __attribute__((ext_vector_type(8)))  float  v8f;

// ---- output layout (f32 elements), reference return order ----
constexpr size_t O_CTXOUT = 0;                       // [B,T,H]
constexpr size_t O_CATTN  = O_CTXOUT + 64ull*64*512; // [B,T,SC]
constexpr size_t O_QATTN  = O_CATTN  + 64ull*64*512; // [B,T,SQ]
constexpr size_t O_CALIGN = O_QATTN  + 64ull*64*128; // [B,H]
constexpr size_t O_QALIGN = O_CALIGN + 64ull*512;    // [B,H]
constexpr size_t O_VP     = O_QALIGN + 64ull*512;    // [B,T]
constexpr size_t O_CQ     = O_VP     + 64ull*64;     // [B,T]
constexpr size_t O_HT     = O_CQ     + 64ull*64;     // [B,H]
constexpr size_t O_CT     = O_HT     + 64ull*512;    // [B,H]

struct Params {
  const float* input; const float* context; const float* question;
  const int* cmask; const int* qmask;
  const float* b_ih; const float* b_hh;
  const float* W_vp; const float* b_vp; const float* W_cq; const float* b_cq;
  const unsigned short* WihB; const unsigned short* WhhB;
  const unsigned short* WcB;  const unsigned short* WqB;
  unsigned short* xA;          // [4 bg][32 ktile][512] bf16 A-tiles of x=[emb|ctx_out]
  unsigned short* hA0;         // [4 bg][16 ktile][512] bf16 A-tiles of h (double buffered)
  unsigned short* hA1;
  unsigned short* cA2;         // [4 bg][16 ktile][512] bf16 A-tiles of ctx align
  unsigned short* qA2;         // same for question align
  float* hF;                   // [64,512] f32 h
  float* ctx0; float* ctx1;    // [64,512] ctx_out double buffer
  float* qO;                   // [64,512] q_out
  unsigned* bar;
  float* out;
};

__device__ __forceinline__ unsigned short f2bf(float f){
  union { float f; unsigned u; } c; c.f = f;
  unsigned u = c.u;
  u += 0x7FFFu + ((u >> 16) & 1u);           // round-to-nearest-even
  return (unsigned short)(u >> 16);
}

// WMMA 16x16x32 bf16 A-matrix lane/vgpr swizzle: element (row r, k kk) -> tile index
__device__ __forceinline__ int aidx(int r, int kk){
  int hi = kk >> 4, k2 = kk & 15, half = k2 >> 3;
  int v = hi*4 + ((k2 & 7) >> 1);
  return (r + half*16)*16 + v*2 + (kk & 1);
}

__device__ __forceinline__ float sigm(float x){ return 1.0f/(1.0f + expf(-x)); }

__device__ __forceinline__ void grid_barrier(unsigned* cnt, unsigned* gen, unsigned nwg){
  __syncthreads();
  if (threadIdx.x == 0){
    __threadfence();
    unsigned g = __hip_atomic_load(gen, __ATOMIC_RELAXED, __HIP_MEMORY_SCOPE_AGENT);
    unsigned a = __hip_atomic_fetch_add(cnt, 1u, __ATOMIC_ACQ_REL, __HIP_MEMORY_SCOPE_AGENT);
    if (a == nwg - 1u){
      __hip_atomic_store(cnt, 0u, __ATOMIC_RELAXED, __HIP_MEMORY_SCOPE_AGENT);
      __hip_atomic_fetch_add(gen, 1u, __ATOMIC_RELEASE, __HIP_MEMORY_SCOPE_AGENT);
    } else {
      while (__hip_atomic_load(gen, __ATOMIC_ACQUIRE, __HIP_MEMORY_SCOPE_AGENT) == g)
        __builtin_amdgcn_s_sleep(2);
    }
    __threadfence();
  }
  __syncthreads();
}

// ---- one-shot: convert weights f32 -> bf16 in WMMA-B-tile-swizzled layout ----
__global__ void prep_kernel(const float* Wih, const float* Whh, const float* Wc, const float* Wq,
                            unsigned short* WihB, unsigned short* WhhB,
                            unsigned short* WcB,  unsigned short* WqB, unsigned* bar)
{
  size_t tid = (size_t)blockIdx.x * blockDim.x + threadIdx.x;
  if (tid == 0){ bar[0] = 0u; bar[1] = 0u; }
  const size_t N1 = 2048ull*1024, N2 = 2048ull*512, N3 = 512ull*1024, N4 = 512ull*1024;
  size_t stride = (size_t)gridDim.x * blockDim.x;
  for (size_t i = tid; i < N1+N2+N3+N4; i += stride){
    const float* src; unsigned short* dst; size_t li; int K;
    if (i < N1)            { src = Wih; dst = WihB; li = i;           K = 1024; }
    else if (i < N1+N2)    { src = Whh; dst = WhhB; li = i - N1;      K = 512;  }
    else if (i < N1+N2+N3) { src = Wc;  dst = WcB;  li = i - N1 - N2; K = 1024; }
    else                   { src = Wq;  dst = WqB;  li = i - N1-N2-N3;K = 1024; }
    int n = (int)(li / K), k = (int)(li % K);
    int nt = n >> 4, r = n & 15, kt = k >> 5, kk = k & 31;
    int off = (r + ((kk & 8) ? 16 : 0))*16 + ((kk & 16) >> 1) + (kk & 7);
    dst[(((size_t)nt*(K >> 5) + kt) << 9) + off] = f2bf(src[li]);
  }
}

__global__ __launch_bounds__(NTHR, 1) void decoder_kernel(Params p)
{
  __shared__ float ldsC[512];                    // persistent LSTM cell state (32 hid x 16 batch)
  __shared__ __align__(16) float ldsS[2048];     // per-phase scratch (8KB)
  const int wg  = blockIdx.x;
  const int tid = threadIdx.x;
  const int bg  = wg >> 4;                       // batch group 0..3 (16 batches)
  const int hs  = wg & 15;                       // hidden slice / n-slice / s-slice 0..15
  const int wave = tid >> 5, lane = tid & 31;
  const size_t ft = (size_t)wg * NTHR + tid;     // 0..16383 flat over grid

  // ---------------- init: zero state buffers, stage emb(t=0) tiles ----------
  ((unsigned*)p.hA0)[ft] = 0u;                                  // 16384 u32 = 64KB
  { int bgi = (int)(ft >> 12); int ii = (int)(ft & 4095);       // xA ctx-half zero
    ((unsigned*)p.xA)[(size_t)bgi*8192 + 4096 + ii] = 0u; }
  p.ctx0[ft] = 0.0f; p.ctx0[ft + 16384] = 0.0f;
  #pragma unroll
  for (int rep = 0; rep < 2; ++rep){
    size_t pp = ft + (size_t)rep*16384;                         // 32768 emb elements
    int b = (int)(pp >> 9), k = (int)(pp & 511);
    int bgi = b >> 4, r = b & 15, kt = k >> 5;
    p.xA[(((size_t)(bgi*32 + kt)) << 9) + aidx(r, k & 31)] =
        f2bf(p.input[(size_t)b*Tv*DINv + k]);
  }
  ldsC[tid] = 0.0f; ldsC[tid + 256] = 0.0f;
  grid_barrier(p.bar, p.bar + 1, NWG);

  for (int t = 0; t < Tv; ++t){
    float* ctxPrev = (t & 1) ? p.ctx1 : p.ctx0;
    float* ctxNew  = (t & 1) ? p.ctx0 : p.ctx1;
    const unsigned short* hAr = (t & 1) ? p.hA1 : p.hA0;  // h_{t-1} tiles (read)
    unsigned short*       hAw = (t & 1) ? p.hA0 : p.hA1;  // h_t tiles (write)

    // ============ Phase A: LSTM gates GEMM (WMMA) + pointwise ============
    {
      const int g  = wave >> 1;                 // gate 0..3 (i,f,g,o)
      const int ns = wave & 1;
      const int n0 = g*512 + hs*32 + ns*16;     // gate column base
      const int nt = n0 >> 4;
      const int ncol = n0 + (lane & 15);
      float bias = p.b_ih[ncol] + p.b_hh[ncol];
      v8f acc;
      #pragma unroll
      for (int i = 0; i < 8; ++i) acc[i] = bias;
      #pragma unroll 4
      for (int kt = 0; kt < 32; ++kt){          // x part, K=1024
        v16bf a  = *(const v16bf*)(p.xA   + (((size_t)(bg*32 + kt)) << 9) + lane*16);
        v16bf bm = *(const v16bf*)(p.WihB + (((size_t)(nt*32 + kt)) << 9) + lane*16);
        acc = __builtin_amdgcn_wmma_f32_16x16x32_bf16(false, a, false, bm, (short)0, acc, false, false);
      }
      #pragma unroll 4
      for (int kt = 0; kt < 16; ++kt){          // h part, K=512
        v16bf a  = *(const v16bf*)(hAr    + (((size_t)(bg*16 + kt)) << 9) + lane*16);
        v16bf bm = *(const v16bf*)(p.WhhB + (((size_t)(nt*16 + kt)) << 9) + lane*16);
        acc = __builtin_amdgcn_wmma_f32_16x16x32_bf16(false, a, false, bm, (short)0, acc, false, false);
      }
      #pragma unroll
      for (int r = 0; r < 8; ++r)               // stash gate tiles: [gate][hid32][batch16]
        ldsS[(g*32 + ns*16 + (lane & 15))*16 + (r + 8*(lane >> 4))] = acc[r];
      __syncthreads();
      for (int pp = tid; pp < 512; pp += NTHR){
        int j = pp >> 4, m = pp & 15;
        float gi = ldsS[ j      *16 + m];
        float gf = ldsS[(32 + j)*16 + m];
        float gg = ldsS[(64 + j)*16 + m];
        float go = ldsS[(96 + j)*16 + m];
        float cOld = ldsC[pp];
        float cNew = sigm(gf)*cOld + sigm(gi)*tanhf(gg);
        ldsC[pp] = cNew;
        float h = sigm(go)*tanhf(cNew);
        int b = bg*16 + m, jg = hs*32 + j;
        p.hF[(size_t)b*Hv + jg] = h;
        hAw[(((size_t)(bg*16 + hs)) << 9) + aidx(m, j)] = f2bf(h);
        if (t == Tv-1){
          p.out[O_HT + (size_t)b*Hv + jg] = h;
          p.out[O_CT + (size_t)b*Hv + jg] = cNew;
        }
      }
    }
    grid_barrier(p.bar, p.bar + 1, NWG);

    // ============ Phase B: dot-attention (per-batch WG) ============
    {
      const int b = wg;
      const float* ctx = p.context  + (size_t)b*SCv*Hv;
      const float* qst = p.question + (size_t)b*SQv*Hv;
      float* hL  = ldsS;         // 512
      float* scL = ldsS + 512;   // 512
      float* red = ldsS + 1024;  // 256
      float* qL  = ldsS + 1280;  // 128
      hL[tid]       = p.hF[(size_t)b*Hv + tid];
      hL[tid + 256] = p.hF[(size_t)b*Hv + tid + 256];
      __syncthreads();
      for (int s = tid; s < SCv; s += NTHR){
        const float* row = ctx + (size_t)s*Hv;
        float acc = 0.f;
        for (int k = 0; k < Hv; k += 4){
          float4 c4 = *(const float4*)(row + k);
          float4 h4 = *(const float4*)(hL + k);
          acc += c4.x*h4.x + c4.y*h4.y + c4.z*h4.z + c4.w*h4.w;
        }
        if (p.cmask[(size_t)b*SCv + s] == 0) acc = -__builtin_inff();
        scL[s] = acc;
      }
      __syncthreads();
      red[tid] = fmaxf(scL[tid], scL[tid + 256]);
      __syncthreads();
      for (int off = 128; off > 0; off >>= 1){
        if (tid < off) red[tid] = fmaxf(red[tid], red[tid + off]);
        __syncthreads();
      }
      float mx = red[0]; __syncthreads();
      float e0 = expf(scL[tid] - mx), e1 = expf(scL[tid + 256] - mx);
      red[tid] = e0 + e1;
      __syncthreads();
      for (int off = 128; off > 0; off >>= 1){
        if (tid < off) red[tid] += red[tid + off];
        __syncthreads();
      }
      float inv = 1.0f / red[0]; __syncthreads();
      float a0 = e0*inv, a1 = e1*inv;
      scL[tid] = a0; scL[tid + 256] = a1;
      p.out[O_CATTN + ((size_t)b*Tv + t)*SCv + tid]       = a0;
      p.out[O_CATTN + ((size_t)b*Tv + t)*SCv + tid + 256] = a1;
      __syncthreads();
      {   // context align (coalesced over k)
        float al0 = 0.f, al1 = 0.f;
        for (int s = 0; s < SCv; ++s){
          float at = scL[s];
          al0 += at * ctx[(size_t)s*Hv + tid];
          al1 += at * ctx[(size_t)s*Hv + tid + 256];
        }
        int r = b & 15, k0 = tid, k1 = tid + 256;
        p.cA2[(((size_t)(bg*16 + (k0 >> 5))) << 9) + aidx(r, k0 & 31)] = f2bf(al0);
        p.cA2[(((size_t)(bg*16 + (k1 >> 5))) << 9) + aidx(r, k1 & 31)] = f2bf(al1);
        if (t == Tv-1){
          p.out[O_CALIGN + (size_t)b*Hv + k0] = al0;
          p.out[O_CALIGN + (size_t)b*Hv + k1] = al1;
        }
      }
      __syncthreads();
      if (tid < SQv){
        const float* row = qst + (size_t)tid*Hv;
        float acc = 0.f;
        for (int k = 0; k < Hv; k += 4){
          float4 c4 = *(const float4*)(row + k);
          float4 h4 = *(const float4*)(hL + k);
          acc += c4.x*h4.x + c4.y*h4.y + c4.z*h4.z + c4.w*h4.w;
        }
        if (p.qmask[(size_t)b*SQv + tid] == 0) acc = -__builtin_inff();
        qL[tid] = acc;
      }
      __syncthreads();
      red[tid] = (tid < SQv) ? qL[tid] : -__builtin_inff();
      __syncthreads();
      for (int off = 128; off > 0; off >>= 1){
        if (tid < off) red[tid] = fmaxf(red[tid], red[tid + off]);
        __syncthreads();
      }
      float qmx = red[0]; __syncthreads();
      float qe = (tid < SQv) ? expf(qL[tid] - qmx) : 0.f;
      red[tid] = qe;
      __syncthreads();
      for (int off = 128; off > 0; off >>= 1){
        if (tid < off) red[tid] += red[tid + off];
        __syncthreads();
      }
      float qinv = 1.0f / red[0]; __syncthreads();
      if (tid < SQv){
        float qa = qe * qinv;
        qL[tid] = qa;
        p.out[O_QATTN + ((size_t)b*Tv + t)*SQv + tid] = qa;
      }
      __syncthreads();
      {   // question align
        float al0 = 0.f, al1 = 0.f;
        for (int s = 0; s < SQv; ++s){
          float at = qL[s];
          al0 += at * qst[(size_t)s*Hv + tid];
          al1 += at * qst[(size_t)s*Hv + tid + 256];
        }
        int r = b & 15, k0 = tid, k1 = tid + 256;
        p.qA2[(((size_t)(bg*16 + (k0 >> 5))) << 9) + aidx(r, k0 & 31)] = f2bf(al0);
        p.qA2[(((size_t)(bg*16 + (k1 >> 5))) << 9) + aidx(r, k1 & 31)] = f2bf(al1);
        if (t == Tv-1){
          p.out[O_QALIGN + (size_t)b*Hv + k0] = al0;
          p.out[O_QALIGN + (size_t)b*Hv + k1] = al1;
        }
      }
      if (t + 1 < Tv){   // stage next step's embedding A-tiles
        for (int k = tid; k < DINv; k += NTHR){
          int r = b & 15, kt = k >> 5;
          p.xA[(((size_t)(bg*32 + kt)) << 9) + aidx(r, k & 31)] =
              f2bf(p.input[((size_t)b*Tv + (t + 1))*DINv + k]);
        }
      }
    }
    grid_barrier(p.bar, p.bar + 1, NWG);

    // ============ Phase D: linear_out GEMMs (WMMA, fused ctx+q, K-split) ============
    {
      const int tIdx  = hs*4 + (wave & 3);       // combined N-tile 0..63 (0..31 ctx, 32..63 q)
      const int khalf = wave >> 2;               // K split across wave pairs
      const bool isCtx = (tIdx < 32);
      const int ntile = isCtx ? tIdx : (tIdx - 32);
      const unsigned short* Wmat = isCtx ? p.WcB : p.WqB;
      const unsigned short* A2   = isCtx ? p.cA2 : p.qA2;
      const unsigned short* Abase = (khalf == 0)
          ? (hAw + (((size_t)(bg*16)) << 9))     // first 512 K = dec (h_t)
          : (A2  + (((size_t)(bg*16)) << 9));    // second 512 K = align
      v8f acc;
      #pragma unroll
      for (int i = 0; i < 8; ++i) acc[i] = 0.f;
      #pragma unroll 4
      for (int kk = 0; kk < 16; ++kk){
        v16bf a  = *(const v16bf*)(Abase + (((size_t)kk) << 9) + lane*16);
        v16bf bm = *(const v16bf*)(Wmat + (((size_t)(ntile*32 + khalf*16 + kk)) << 9) + lane*16);
        acc = __builtin_amdgcn_wmma_f32_16x16x32_bf16(false, a, false, bm, (short)0, acc, false, false);
      }
      if (wave < 4){
        #pragma unroll
        for (int r = 0; r < 8; ++r) ldsS[wave*256 + r*32 + lane] = acc[r];
      }
      __syncthreads();
      if (wave >= 4){
        #pragma unroll
        for (int r = 0; r < 8; ++r) ldsS[(wave - 4)*256 + r*32 + lane] += acc[r];
      }
      __syncthreads();
      for (int pp = tid; pp < 1024; pp += NTHR){
        int tl = pp >> 8, r = (pp >> 5) & 7, l = pp & 31;
        int m = r + 8*(l >> 4), nn = l & 15;
        int ti = hs*4 + tl;
        int b = bg*16 + m;
        float val = tanhf(ldsS[pp]);
        if (ti < 32){
          int n = ti*16 + nn;
          ctxNew[(size_t)b*Hv + n] = val;
          p.out[O_CTXOUT + ((size_t)b*Tv + t)*Hv + n] = val;
          int xk = 512 + n;    // input-feed half of x for next step
          p.xA[(((size_t)(bg*32 + (xk >> 5))) << 9) + aidx(m, xk & 31)] = f2bf(val);
        } else {
          int n = (ti - 32)*16 + nn;
          p.qO[(size_t)b*Hv + n] = val;
        }
      }
    }
    grid_barrier(p.bar, p.bar + 1, NWG);

    // ============ Phase E: vp / cq switch gates ============
    {
      const int b = wg;
      float avp = 0.f, acq = 0.f;
      for (int i = tid; i < 2048; i += NTHR){
        float vv, vc;
        if (i < 512){ float d = p.hF[(size_t)b*Hv + i]; vv = d; vc = d; }
        else if (i < 1024){
          vv = ctxNew[(size_t)b*Hv + (i - 512)];
          vc = p.qO [(size_t)b*Hv + (i - 512)];
        } else if (i < 1536){
          float x = p.input[((size_t)b*Tv + t)*DINv + (i - 1024)]; vv = x; vc = x;
        } else {
          float x = ctxPrev[(size_t)b*Hv + (i - 1536)]; vv = x; vc = x;
        }
        avp += p.W_vp[i]*vv;
        acq += p.W_cq[i]*vc;
      }
      ldsS[tid] = avp; ldsS[256 + tid] = acq;
      __syncthreads();
      for (int off = 128; off > 0; off >>= 1){
        if (tid < off){ ldsS[tid] += ldsS[tid + off]; ldsS[256 + tid] += ldsS[256 + tid + off]; }
        __syncthreads();
      }
      if (tid == 0){
        p.out[O_VP + (size_t)b*Tv + t] = sigm(ldsS[0]   + p.b_vp[0]);
        p.out[O_CQ + (size_t)b*Tv + t] = sigm(ldsS[256] + p.b_cq[0]);
      }
    }
    grid_barrier(p.bar, p.bar + 1, NWG);
  }
}

extern "C" void kernel_launch(void* const* d_in, const int* in_sizes, int n_in,
                              void* d_out, int out_size, void* d_ws, size_t ws_size,
                              hipStream_t stream)
{
  (void)in_sizes; (void)n_in; (void)out_size; (void)ws_size;
  char* ws = (char*)d_ws;
  size_t off = 0;
  auto take = [&](size_t bytes) -> char* {
    char* r = ws + off; off = (off + bytes + 255) & ~(size_t)255; return r;
  };
  unsigned short* WihB = (unsigned short*)take(2048ull*1024*2);
  unsigned short* WhhB = (unsigned short*)take(2048ull*512*2);
  unsigned short* WcB  = (unsigned short*)take(512ull*1024*2);
  unsigned short* WqB  = (unsigned short*)take(512ull*1024*2);
  unsigned short* xA   = (unsigned short*)take(4ull*32*512*2);
  unsigned short* hA0  = (unsigned short*)take(4ull*16*512*2);
  unsigned short* hA1  = (unsigned short*)take(4ull*16*512*2);
  unsigned short* cA2  = (unsigned short*)take(4ull*16*512*2);
  unsigned short* qA2  = (unsigned short*)take(4ull*16*512*2);
  float* hF   = (float*)take(64ull*512*4);
  float* ctx0 = (float*)take(64ull*512*4);
  float* ctx1 = (float*)take(64ull*512*4);
  float* qO   = (float*)take(64ull*512*4);
  unsigned* bar = (unsigned*)take(256);

  prep_kernel<<<dim3(2048), dim3(256), 0, stream>>>(
      (const float*)d_in[5], (const float*)d_in[6],
      (const float*)d_in[9], (const float*)d_in[10],
      WihB, WhhB, WcB, WqB, bar);

  Params prm;
  prm.input    = (const float*)d_in[0];
  prm.context  = (const float*)d_in[1];
  prm.question = (const float*)d_in[2];
  prm.cmask    = (const int*)d_in[3];
  prm.qmask    = (const int*)d_in[4];
  prm.b_ih     = (const float*)d_in[7];
  prm.b_hh     = (const float*)d_in[8];
  prm.W_vp     = (const float*)d_in[11];
  prm.b_vp     = (const float*)d_in[12];
  prm.W_cq     = (const float*)d_in[13];
  prm.b_cq     = (const float*)d_in[14];
  prm.WihB = WihB; prm.WhhB = WhhB; prm.WcB = WcB; prm.WqB = WqB;
  prm.xA = xA; prm.hA0 = hA0; prm.hA1 = hA1; prm.cA2 = cA2; prm.qA2 = qA2;
  prm.hF = hF; prm.ctx0 = ctx0; prm.ctx1 = ctx1; prm.qO = qO;
  prm.bar = bar;
  prm.out = (float*)d_out;

  decoder_kernel<<<dim3(NWG), dim3(NTHR), 0, stream>>>(prm);
}
